// GNN_GRU_21655225106539
// MI455X (gfx1250) — matrix operand order repaired and checked by reading, not compiled, not run on hardware
//
#include <hip/hip_runtime.h>
#include <hip/hip_fp16.h>

#define H 128

typedef __attribute__((ext_vector_type(16))) _Float16 v16h;
typedef __attribute__((ext_vector_type(8)))  _Float16 h8;
typedef __attribute__((ext_vector_type(8)))  float    v8f;
typedef __attribute__((ext_vector_type(4)))  int      v4i;

// ---------------------------------------------------------------------------
// CDNA5 async global->LDS staging when the toolchain exposes it (ASYNCcnt
// path, CDNA5 ISA); falls back to b128 load + ds store.
// Probe-learned prototype: (v4i addrspace(1)*, v4i addrspace(3)*, imm, imm).
// ---------------------------------------------------------------------------
#if __has_builtin(__builtin_amdgcn_global_load_async_to_lds_b128)
#define ASYNC_LDS 1
typedef __attribute__((address_space(1))) v4i gv4i;
typedef __attribute__((address_space(3))) v4i lv4i;
#else
#define ASYNC_LDS 0
#endif

__device__ __forceinline__ void lds_cp16(const _Float16* gp, _Float16* lp)
{
#if ASYNC_LDS
    __builtin_amdgcn_global_load_async_to_lds_b128(
        (gv4i*)(unsigned long long)gp,
        (lv4i*)(unsigned int)(unsigned long long)lp,
        0, 0);
#else
    *(h8*)lp = *(const h8*)gp;
#endif
}

__device__ __forceinline__ void lds_cp_wait()
{
#if ASYNC_LDS
#if __has_builtin(__builtin_amdgcn_s_wait_asynccnt)
    __builtin_amdgcn_s_wait_asynccnt(0);
#else
    asm volatile("s_wait_asynccnt 0" ::: "memory");
#endif
#endif
}

// ---------------------------------------------------------------------------
// WMMA fragment helpers (wave32 layouts per CDNA5 ISA 7.12.2).
// A fragment: lane (r,kh): halves j = K kh*16 + j (matches B fragment order)
// B fragment: lane (n,kh): halves j = K kh*16 + j, contiguous in Wt[N][K].
// ---------------------------------------------------------------------------
__device__ __forceinline__ v16h load_afrag(const _Float16* arow, int kh)
{
    h8 alo = *(const h8*)(arow + kh * 16);
    h8 ahi = *(const h8*)(arow + kh * 16 + 8);
    v16h a;
#pragma unroll
    for (int i = 0; i < 8; ++i) { a[i] = alo[i]; a[i + 8] = ahi[i]; }
    return a;
}

__device__ __forceinline__ v16h load_bfrag(const _Float16* wp_kb, int kh)
{
    h8 blo = *(const h8*)(wp_kb + kh * 16);
    h8 bhi = *(const h8*)(wp_kb + kh * 16 + 8);
    v16h b;
#pragma unroll
    for (int i = 0; i < 8; ++i) { b[i] = blo[i]; b[i + 8] = bhi[i]; }
    return b;
}

__device__ __forceinline__ void wmma_epilogue(v8f c, int mbase, int M, int col,
                                              int ldo, const float* bias, int relu,
                                              float* outf, _Float16* outh)
{
    int kh = threadIdx.x >> 4;
    float bv = bias ? bias[col] : 0.0f;
#pragma unroll
    for (int i = 0; i < 8; ++i) {
        int m = mbase + kh * 8 + i;            // C layout: VGPR i -> M = kh*8+i
        if (m >= M) continue;
        float v = c[i] + bv;
        if (relu) v = fmaxf(v, 0.0f);
        size_t off = (size_t)m * ldo + col;
        if (outf) outf[off] = v;
        if (outh) outh[off] = (_Float16)v;
    }
}

#define LDS_STRIDE 72   // 32x64 f16 tile, padded rows (144B, 16B-aligned)

// ---- GEMM, A = f16 [M,K]; Wt = f16 [Nout][K]; 32x128 tile, K-step 64 ------
__global__ __launch_bounds__(256)
void gemm_f16A(const _Float16* __restrict__ A, int K, int M,
               const _Float16* __restrict__ Wt,
               const float* __restrict__ bias,
               float* outf, _Float16* outh, int ldo, int relu)
{
    __shared__ __align__(16) _Float16 sA[32 * LDS_STRIDE];
    int mbase = blockIdx.x * 32;
    int nblock = blockIdx.y * 128;
    int lane = threadIdx.x, wave = threadIdx.y;
    int t = wave * 32 + lane;
    int col = nblock + wave * 16 + (lane & 15);
    const _Float16* wp = Wt + (size_t)col * K;
    // staging: thread t copies 16B: row lr, halves lk..lk+7
    int lr = t >> 3, lk = (t & 7) * 8;
    int lrow = mbase + lr; if (lrow >= M) lrow = M - 1;
    const _Float16* gsrc = A + (size_t)lrow * K + lk;
    _Float16* ldst = &sA[lr * LDS_STRIDE + lk];
    int r = lane & 15, kh = lane >> 4;
    const _Float16* arow0 = &sA[r * LDS_STRIDE];
    const _Float16* arow1 = &sA[(16 + r) * LDS_STRIDE];
    v8f c0 = {}, c1 = {};
    for (int kb = 0; kb < K; kb += 64) {
        __syncthreads();
        lds_cp16(gsrc + kb, ldst);
        lds_cp_wait();
        __syncthreads();
#pragma unroll
        for (int ks = 0; ks < 64; ks += 32) {
            v16h b  = load_bfrag(wp + kb + ks, kh);
            v16h a0 = load_afrag(arow0 + ks, kh);
            v16h a1 = load_afrag(arow1 + ks, kh);
            c0 = __builtin_amdgcn_wmma_f32_16x16x32_f16(false, a0, false, b,
                                                        (short)0, c0, false, false);
            c1 = __builtin_amdgcn_wmma_f32_16x16x32_f16(false, a1, false, b,
                                                        (short)0, c1, false, false);
        }
    }
    wmma_epilogue(c0, mbase,      M, col, ldo, bias, relu, outf, outh);
    wmma_epilogue(c1, mbase + 16, M, col, ldo, bias, relu, outf, outh);
}

// ---- edge-update layer 1: A = cat[x[src], x[dst], e, u[eb]], K=512 fused ---
__global__ __launch_bounds__(256)
void gemm_catA(const _Float16* __restrict__ xh, const _Float16* __restrict__ eh,
               const _Float16* __restrict__ uh,
               const int* __restrict__ srci, const int* __restrict__ dsti,
               const int* __restrict__ ebat,
               const _Float16* __restrict__ Wt,      // [128][512]
               const float* __restrict__ bias,
               _Float16* outh, int relu)
{
    __shared__ __align__(16) _Float16 sA[32 * LDS_STRIDE];
    __shared__ const _Float16* rowp[4][32];
    int mbase = blockIdx.x * 32;                      // E % 32 == 0
    int lane = threadIdx.x, wave = threadIdx.y;
    int t = wave * 32 + lane;
    if (t < 32) {
        int row = mbase + t;
        rowp[0][t] = xh + (size_t)srci[row] * H;
        rowp[1][t] = xh + (size_t)dsti[row] * H;
        rowp[2][t] = eh + (size_t)row * H;
        rowp[3][t] = uh + (size_t)ebat[row] * H;
    }
    int col = wave * 16 + (lane & 15);
    const _Float16* wp = Wt + (size_t)col * 512;
    int lr = t >> 3, lk = (t & 7) * 8;
    _Float16* ldst = &sA[lr * LDS_STRIDE + lk];
    int r = lane & 15, kh = lane >> 4;
    const _Float16* arow0 = &sA[r * LDS_STRIDE];
    const _Float16* arow1 = &sA[(16 + r) * LDS_STRIDE];
    v8f c0 = {}, c1 = {};
    for (int kb = 0; kb < 512; kb += 64) {
        __syncthreads();                              // also covers rowp init
        int seg = kb >> 7, kin = kb & 127;            // 64-block never straddles
        lds_cp16(rowp[seg][lr] + kin + lk, ldst);
        lds_cp_wait();
        __syncthreads();
#pragma unroll
        for (int ks = 0; ks < 64; ks += 32) {
            v16h b  = load_bfrag(wp + kb + ks, kh);
            v16h a0 = load_afrag(arow0 + ks, kh);
            v16h a1 = load_afrag(arow1 + ks, kh);
            c0 = __builtin_amdgcn_wmma_f32_16x16x32_f16(false, a0, false, b,
                                                        (short)0, c0, false, false);
            c1 = __builtin_amdgcn_wmma_f32_16x16x32_f16(false, a1, false, b,
                                                        (short)0, c1, false, false);
        }
    }
    wmma_epilogue(c0, mbase,      1 << 30, col, H, bias, relu, nullptr, outh);
    wmma_epilogue(c1, mbase + 16, 1 << 30, col, H, bias, relu, nullptr, outh);
}

// ---- GEMM, A = f32 [M,Kact] (encoders only, small K), 16x128 tile ----------
__global__ __launch_bounds__(256)
void gemm_f32A(const float* __restrict__ A, int lda, int Kact,
               const _Float16* __restrict__ Wt, int Kpad,
               const float* __restrict__ bias,
               float* outf, _Float16* outh, int relu)
{
    __shared__ __align__(16) _Float16 sA[16 * 32];
    int mbase = blockIdx.x * 16;
    int lane = threadIdx.x, wave = threadIdx.y;
    int t = wave * 32 + lane;
    int col = wave * 16 + (lane & 15);
    const _Float16* wp = Wt + (size_t)col * Kpad;
    int r = lane & 15, kh = lane >> 4;
    v8f c = {};
    for (int kb = 0; kb < Kpad; kb += 32) {
        __syncthreads();
#pragma unroll
        for (int e2 = 0; e2 < 2; ++e2) {
            int idx = t + e2 * 256;
            int rr = idx >> 5, kk = idx & 31;
            int kg = kb + kk;
            sA[idx] = (_Float16)(kg < Kact ? A[(size_t)(mbase + rr) * lda + kg] : 0.0f);
        }
        __syncthreads();
        v16h b = load_bfrag(wp + kb, kh);
        v16h a = load_afrag(&sA[r * 32], kh);
        c = __builtin_amdgcn_wmma_f32_16x16x32_f16(false, a, false, b,
                                                   (short)0, c, false, false);
    }
    wmma_epilogue(c, mbase, 1 << 30, col, H, bias, relu, outf, outh);
}

// ---- LayerNorm over H=128 (one row per block), optional relu + segment-sum --
__global__ __launch_bounds__(128)
void ln_kernel(const _Float16* __restrict__ in,
               const float* __restrict__ g, const float* __restrict__ b,
               _Float16* outh, float* outf, int relu,
               float* __restrict__ agg, const int* __restrict__ batch)
{
    __shared__ float red[128];
    __shared__ float stats[2];
    int row = blockIdx.x, t = threadIdx.x;
    float v = (float)in[(size_t)row * H + t];
    red[t] = v; __syncthreads();
    for (int s = 64; s > 0; s >>= 1) { if (t < s) red[t] += red[t + s]; __syncthreads(); }
    if (t == 0) stats[0] = red[0] * (1.0f / H);
    __syncthreads();
    float mean = stats[0];
    float d = v - mean;
    red[t] = d * d; __syncthreads();
    for (int s = 64; s > 0; s >>= 1) { if (t < s) red[t] += red[t + s]; __syncthreads(); }
    if (t == 0) stats[1] = red[0] * (1.0f / H);
    __syncthreads();
    float y = d * rsqrtf(stats[1] + 1e-5f) * g[t] + b[t];
    if (relu) y = fmaxf(y, 0.0f);
    if (outh) outh[(size_t)row * H + t] = (_Float16)y;
    if (outf) outf[(size_t)row * H + t] = y;
    if (agg)  atomicAdd(&agg[(size_t)batch[row] * H + t], y);
}

// ---- attention scores: s = leaky_relu(q[dst]+k+gu[nb[dst]]) . a ; seg-max ---
__global__ __launch_bounds__(256)
void attn_score(const float* __restrict__ q, const _Float16* __restrict__ kh,
                const float* __restrict__ gu, const float* __restrict__ atta,
                const int* __restrict__ dsti, const int* __restrict__ nbat,
                float* __restrict__ s, unsigned int* __restrict__ smax, int E)
{
    int eid = blockIdx.x * 8 + threadIdx.y;
    if (eid >= E) return;
    int lane = threadIdx.x;
    int d = dsti[eid];
    int gi = nbat[d];
    float acc = 0.0f;
#pragma unroll
    for (int j = 0; j < 4; ++j) {
        int c = lane + j * 32;
        float v = q[(size_t)d * H + c] + (float)kh[(size_t)eid * H + c]
                + gu[(size_t)gi * H + c];
        v = v > 0.0f ? v : 0.2f * v;
        acc += v * atta[c];
    }
    for (int m = 16; m > 0; m >>= 1) acc += __shfl_xor(acc, m, 32);
    if (lane == 0) {
        s[eid] = acc;
        unsigned int u = __float_as_uint(acc);
        u = (u >> 31) ? ~u : (u | 0x80000000u);   // monotonic float->uint
        atomicMax(&smax[d], u);
    }
}

__global__ __launch_bounds__(256)
void attn_softmax(float* __restrict__ s, const unsigned int* __restrict__ smax,
                  float* __restrict__ denom, const int* __restrict__ dsti, int E)
{
    int eid = blockIdx.x * 256 + threadIdx.x;
    if (eid >= E) return;
    int d = dsti[eid];
    unsigned int u = smax[d];
    float mx = __uint_as_float((u & 0x80000000u) ? (u ^ 0x80000000u) : ~u);
    float ex = expf(s[eid] - mx);
    s[eid] = ex;
    atomicAdd(&denom[d], ex);
}

__global__ __launch_bounds__(256)
void attn_aggregate(const float* __restrict__ ex, const float* __restrict__ denom,
                    const _Float16* __restrict__ kh, const int* __restrict__ dsti,
                    float* __restrict__ xacc, int total)   // total = E*H
{
    int idx = blockIdx.x * 256 + threadIdx.x;
    if (idx >= total) return;
    int eid = idx >> 7, c = idx & 127;
    int d = dsti[eid];
    float alpha = ex[eid] / (denom[d] + 1e-16f);
    atomicAdd(&xacc[(size_t)d * H + c], alpha * (float)kh[idx]);
}

__global__ __launch_bounds__(256)
void node_finalize(const float* __restrict__ xacc, _Float16* __restrict__ xh,
                   float* __restrict__ nagg, const int* __restrict__ nbat, int total)
{
    int idx = blockIdx.x * 256 + threadIdx.x;
    if (idx >= total) return;
    int row = idx >> 7, c = idx & 127;
    float v = fmaxf(xacc[idx], 0.0f);
    xh[idx] = (_Float16)v;
    atomicAdd(&nagg[(size_t)nbat[row] * H + c], v);
}

__global__ __launch_bounds__(256)
void build_ucat(const float* __restrict__ u1, const float* __restrict__ na,
                const float* __restrict__ ea, _Float16* __restrict__ ucat, int total)
{
    int idx = blockIdx.x * 256 + threadIdx.x;
    if (idx >= total) return;
    int g = idx / 384, c = idx % 384;
    float v = (c < 128) ? u1[g * 128 + c]
            : (c < 256) ? na[g * 128 + (c - 128)]
                        : ea[g * 128 + (c - 256)];
    ucat[idx] = (_Float16)v;
}

// GRU step with h_prev = 0:  h = (1 - sigmoid(gi_z)) * tanh(gi_n)
__global__ __launch_bounds__(256)
void gru_elem(const float* __restrict__ gi, float* __restrict__ hout,
              _Float16* __restrict__ hh, int total)
{
    int idx = blockIdx.x * 256 + threadIdx.x;
    if (idx >= total) return;
    int g = idx >> 7, c = idx & 127;
    float z = 1.0f / (1.0f + expf(-gi[g * 384 + 128 + c]));
    float n = tanhf(gi[g * 384 + 256 + c]);
    float h = (1.0f - z) * n;
    hout[idx] = h;
    if (hh) hh[idx] = (_Float16)h;
}

// ---- weight conversion: W[K,N] f32 -> Wt[N,Kpad] f16 (zero padded) ---------
__global__ __launch_bounds__(256)
void conv_wT(const float* __restrict__ W, _Float16* __restrict__ Wt,
             int K, int N, int Kpad)
{
    int idx = blockIdx.x * 256 + threadIdx.x;
    if (idx >= N * Kpad) return;
    int n = idx / Kpad, k = idx % Kpad;
    Wt[idx] = (_Float16)(k < K ? W[(size_t)k * N + n] : 0.0f);
}

__global__ __launch_bounds__(256)
void conv_copy(const float* __restrict__ W, _Float16* __restrict__ Wt, int count)
{
    int idx = blockIdx.x * 256 + threadIdx.x;
    if (idx < count) Wt[idx] = (_Float16)W[idx];
}

// ===========================================================================
extern "C" void kernel_launch(void* const* d_in, const int* in_sizes, int n_in,
                              void* d_out, int out_size, void* d_ws, size_t ws_size,
                              hipStream_t stream)
{
    const float* node_x   = (const float*)d_in[0];
    const float* edge_att = (const float*)d_in[1];
    const float* glob_x   = (const float*)d_in[2];
    const int*   eidx     = (const int*)d_in[3];
    const int*   nbat     = (const int*)d_in[4];
    const int*   ebat     = (const int*)d_in[5];
    const float *ne_w1=(const float*)d_in[6],  *ne_b1=(const float*)d_in[7],
                *ne_w2=(const float*)d_in[8],  *ne_b2=(const float*)d_in[9];
    const float *ee_w1=(const float*)d_in[10], *ee_b1=(const float*)d_in[11],
                *ee_w2=(const float*)d_in[12], *ee_b2=(const float*)d_in[13];
    const float *ge_w1=(const float*)d_in[14], *ge_b1=(const float*)d_in[15],
                *ge_w2=(const float*)d_in[16], *ge_b2=(const float*)d_in[17];
    const float *eu_w1=(const float*)d_in[18], *eu_b1=(const float*)d_in[19],
                *eu_w2=(const float*)d_in[20], *eu_b2=(const float*)d_in[21],
                *eu_g =(const float*)d_in[22], *eu_b =(const float*)d_in[23];
    const float *wq=(const float*)d_in[24], *we=(const float*)d_in[25],
                *wu=(const float*)d_in[26], *att_a=(const float*)d_in[27];
    const float *g1w1=(const float*)d_in[28], *g1b1=(const float*)d_in[29],
                *g1w2=(const float*)d_in[30], *g1b2=(const float*)d_in[31],
                *g1g =(const float*)d_in[32], *g1b =(const float*)d_in[33];
    const float *g2w1=(const float*)d_in[34], *g2b1=(const float*)d_in[35],
                *g2w2=(const float*)d_in[36], *g2b2=(const float*)d_in[37],
                *g2g =(const float*)d_in[38], *g2b =(const float*)d_in[39];
    const float *gru_wih=(const float*)d_in[40];
    const float *asd_w=(const float*)d_in[42], *asd_b=(const float*)d_in[43];

    const int N = in_sizes[0] / 64;
    const int E = in_sizes[1] / 16;
    const int G = in_sizes[2] / 32;
    const int* src_i = eidx;
    const int* dst_i = eidx + E;
    float* out = (float*)d_out;

    // ---------------- workspace arena ----------------
    char* base = (char*)d_ws; size_t off = 0;
    auto alloc = [&](size_t bytes) -> void* {
        void* p = base + off; off = (off + bytes + 255) & ~(size_t)255; return p;
    };
    _Float16* ne1t = (_Float16*)alloc(128 * 64  * 2);
    _Float16* ne2t = (_Float16*)alloc(128 * 128 * 2);
    _Float16* ee1t = (_Float16*)alloc(128 * 32  * 2);
    _Float16* ee2t = (_Float16*)alloc(128 * 128 * 2);
    _Float16* ge1t = (_Float16*)alloc(128 * 32  * 2);
    _Float16* ge2t = (_Float16*)alloc(128 * 128 * 2);
    _Float16* eu1t = (_Float16*)alloc(128 * 512 * 2);
    _Float16* eu2t = (_Float16*)alloc(128 * 128 * 2);
    _Float16* wqt  = (_Float16*)alloc(128 * 128 * 2);
    _Float16* wet  = (_Float16*)alloc(128 * 128 * 2);
    _Float16* wut  = (_Float16*)alloc(128 * 128 * 2);
    _Float16* g11t = (_Float16*)alloc(128 * 128 * 2);
    _Float16* g12t = (_Float16*)alloc(128 * 128 * 2);
    _Float16* g21t = (_Float16*)alloc(128 * 384 * 2);
    _Float16* g22t = (_Float16*)alloc(128 * 128 * 2);
    _Float16* gr0t = (_Float16*)alloc(384 * 128 * 2);
    _Float16* gr1t = (_Float16*)alloc(384 * 128 * 2);
    _Float16* asdt = (_Float16*)alloc(128 * 128 * 2);

    _Float16* x_h   = (_Float16*)alloc((size_t)N * H * 2);
    _Float16* e_h   = (_Float16*)alloc((size_t)E * H * 2);
    _Float16* tmp_h = (_Float16*)alloc((size_t)E * H * 2);   // MLP hiddens + k
    float*    q     = (float*)alloc((size_t)N * H * 4);
    float*    gu    = (float*)alloc((size_t)G * H * 4);
    float*    x_acc = (float*)alloc((size_t)N * H * 4);
    float*    s_buf = (float*)alloc((size_t)E * 4);
    unsigned int* smax = (unsigned int*)alloc((size_t)N * 4);
    float*    denom = (float*)alloc((size_t)N * 4);
    float*    nagg  = (float*)alloc((size_t)G * H * 4);
    float*    eagg  = (float*)alloc((size_t)G * H * 4);
    _Float16* u_h   = (_Float16*)alloc((size_t)G * H * 2);
    _Float16* g1h   = (_Float16*)alloc((size_t)G * H * 2);
    _Float16* uAh   = (_Float16*)alloc((size_t)G * H * 2);
    float*    u1    = (float*)alloc((size_t)G * H * 4);
    _Float16* ucat  = (_Float16*)alloc((size_t)G * 384 * 2);
    _Float16* g2h   = (_Float16*)alloc((size_t)G * H * 2);
    _Float16* uBh   = (_Float16*)alloc((size_t)G * H * 2);
    float*    gi1   = (float*)alloc((size_t)G * 384 * 4);
    float*    gi2   = (float*)alloc((size_t)G * 384 * 4);
    _Float16* h1h   = (_Float16*)alloc((size_t)G * H * 2);

    dim3 GB(32, 8);                 // 8 wave32s per GEMM block
    dim3 SB(32, 8);                 // attn_score: 8 edges per block
    auto c1  = [](int n) { return dim3((n + 255) / 256); };
    auto m32 = [](int m) { return dim3((m + 31) / 32); };

    // ---------------- weight conversion ----------------
    conv_wT<<<c1(128*64),  256, 0, stream>>>(ne_w1, ne1t, 64, 128, 64);
    conv_wT<<<c1(128*128), 256, 0, stream>>>(ne_w2, ne2t, 128, 128, 128);
    conv_wT<<<c1(128*32),  256, 0, stream>>>(ee_w1, ee1t, 16, 128, 32);
    conv_wT<<<c1(128*128), 256, 0, stream>>>(ee_w2, ee2t, 128, 128, 128);
    conv_wT<<<c1(128*32),  256, 0, stream>>>(ge_w1, ge1t, 32, 128, 32);
    conv_wT<<<c1(128*128), 256, 0, stream>>>(ge_w2, ge2t, 128, 128, 128);
    conv_wT<<<c1(128*512), 256, 0, stream>>>(eu_w1, eu1t, 512, 128, 512);
    conv_wT<<<c1(128*128), 256, 0, stream>>>(eu_w2, eu2t, 128, 128, 128);
    conv_wT<<<c1(128*128), 256, 0, stream>>>(wq,    wqt,  128, 128, 128);
    conv_wT<<<c1(128*128), 256, 0, stream>>>(we,    wet,  128, 128, 128);
    conv_wT<<<c1(128*128), 256, 0, stream>>>(wu,    wut,  128, 128, 128);
    conv_wT<<<c1(128*128), 256, 0, stream>>>(g1w1,  g11t, 128, 128, 128);
    conv_wT<<<c1(128*128), 256, 0, stream>>>(g1w2,  g12t, 128, 128, 128);
    conv_wT<<<c1(128*384), 256, 0, stream>>>(g2w1,  g21t, 384, 128, 384);
    conv_wT<<<c1(128*128), 256, 0, stream>>>(g2w2,  g22t, 128, 128, 128);
    conv_copy<<<c1(384*128), 256, 0, stream>>>(gru_wih,           gr0t, 384*128);
    conv_copy<<<c1(384*128), 256, 0, stream>>>(gru_wih + 384*128, gr1t, 384*128);
    conv_wT<<<c1(128*128), 256, 0, stream>>>(asd_w, asdt, 128, 128, 128);

    // ---------------- encoders ----------------
    gemm_f32A<<<dim3(N/16), GB, 0, stream>>>(node_x, 64, 64, ne1t, 64, ne_b1,
                                             nullptr, tmp_h, 1);
    gemm_f16A<<<m32(N), GB, 0, stream>>>(tmp_h, 128, N, ne2t, ne_b2,
                                         nullptr, x_h, H, 0);
    gemm_f32A<<<dim3(E/16), GB, 0, stream>>>(edge_att, 16, 16, ee1t, 32, ee_b1,
                                             nullptr, tmp_h, 1);
    gemm_f16A<<<m32(E), GB, 0, stream>>>(tmp_h, 128, E, ee2t, ee_b2,
                                         nullptr, e_h, H, 0);
    gemm_f32A<<<dim3(G/16), GB, 0, stream>>>(glob_x, 32, 32, ge1t, 32, ge_b1,
                                             nullptr, g1h, 1);
    gemm_f16A<<<m32(G), GB, 0, stream>>>(g1h, 128, G, ge2t, ge_b2,
                                         nullptr, u_h, H, 0);

    // ---------------- GNN rounds ----------------
    for (int r = 0; r < 3; ++r) {
        // edge update: tmp = relu(cat @ W1 + b1); e = relu(LN(tmp @ W2 + b2))
        gemm_catA<<<dim3(E/32), GB, 0, stream>>>(x_h, e_h, u_h, src_i, dst_i,
                                                 ebat, eu1t, eu_b1, tmp_h, 1);
        gemm_f16A<<<m32(E), GB, 0, stream>>>(tmp_h, 128, E, eu2t, eu_b2,
                                             nullptr, e_h, H, 0);
        (void)hipMemsetAsync(eagg, 0, (size_t)G * H * 4, stream);
        ln_kernel<<<dim3(E), 128, 0, stream>>>(e_h, eu_g, eu_b, e_h, nullptr, 1,
                                               eagg, ebat);
        // attention projections (q,gu from pre-update x,u; k from new e)
        gemm_f16A<<<m32(N), GB, 0, stream>>>(x_h, 128, N, wqt, nullptr,
                                             q, nullptr, H, 0);
        gemm_f16A<<<m32(G), GB, 0, stream>>>(u_h, 128, G, wut, nullptr,
                                             gu, nullptr, H, 0);
        gemm_f16A<<<m32(E), GB, 0, stream>>>(e_h, 128, E, wet, nullptr,
                                             nullptr, tmp_h, H, 0);   // k
        // segment softmax + aggregation
        (void)hipMemsetAsync(smax, 0, (size_t)N * 4, stream);
        (void)hipMemsetAsync(denom, 0, (size_t)N * 4, stream);
        (void)hipMemsetAsync(x_acc, 0, (size_t)N * H * 4, stream);
        attn_score<<<dim3((E + 7) / 8), SB, 0, stream>>>(q, tmp_h, gu, att_a,
                                                         dst_i, nbat, s_buf, smax, E);
        attn_softmax<<<c1(E), 256, 0, stream>>>(s_buf, smax, denom, dst_i, E);
        attn_aggregate<<<c1(E * H), 256, 0, stream>>>(s_buf, denom, tmp_h, dst_i,
                                                      x_acc, E * H);
        (void)hipMemsetAsync(nagg, 0, (size_t)G * H * 4, stream);
        node_finalize<<<c1(N * H), 256, 0, stream>>>(x_acc, x_h, nagg, nbat, N * H);
        // global update
        gemm_f16A<<<m32(G), GB, 0, stream>>>(u_h, 128, G, g11t, g1b1,
                                             nullptr, g1h, H, 1);
        gemm_f16A<<<m32(G), GB, 0, stream>>>(g1h, 128, G, g12t, g1b2,
                                             nullptr, uAh, H, 0);
        ln_kernel<<<dim3(G), 128, 0, stream>>>(uAh, g1g, g1b, nullptr, u1, 0,
                                               nullptr, nullptr);
        build_ucat<<<c1(G * 384), 256, 0, stream>>>(u1, nagg, eagg, ucat, G * 384);
        gemm_f16A<<<m32(G), GB, 0, stream>>>(ucat, 384, G, g21t, g2b1,
                                             nullptr, g2h, H, 1);
        gemm_f16A<<<m32(G), GB, 0, stream>>>(g2h, 128, G, g22t, g2b2,
                                             nullptr, uBh, H, 0);
        ln_kernel<<<dim3(G), 128, 0, stream>>>(uBh, g2g, g2b, u_h, nullptr, 1,
                                               nullptr, nullptr);
    }

    // ---------------- GRU (h0 = 0) + decoder ----------------
    float* hid = out + (size_t)N * H;                 // [2,G,H]
    gemm_f16A<<<dim3((G + 31) / 32, 3), GB, 0, stream>>>(u_h, 128, G, gr0t, nullptr,
                                                         gi1, nullptr, 384, 0);
    gru_elem<<<c1(G * H), 256, 0, stream>>>(gi1, hid, h1h, G * H);
    gemm_f16A<<<dim3((G + 31) / 32, 3), GB, 0, stream>>>(h1h, 128, G, gr1t, nullptr,
                                                         gi2, nullptr, 384, 0);
    gru_elem<<<c1(G * H), 256, 0, stream>>>(gi2, hid + (size_t)G * H, nullptr, G * H);

    gemm_f16A<<<m32(N), GB, 0, stream>>>(x_h, 128, N, asdt, asd_b,
                                         out, nullptr, H, 0);
    (void)n_in; (void)out_size; (void)ws_size;
}
